// MultiHeadMixer_76407468196393
// MI455X (gfx1250) — matrix operand order repaired and checked by reading, not compiled, not run on hardware
//
#include <hip/hip_runtime.h>
#include <hip/hip_bf16.h>

// Problem constants (from reference): B=8, E=1024, S=2048, H=16, D=64
#define Bb    8
#define Ee    1024
#define Ss    2048
#define Hh    16
#define Dd    64
#define ROWS  (Bb * Dd)   // 512 rows in the (B*D, S) A matrix per head
#define GUARD 160         // zero pad in front of weight row: handles j-k < 0 without masking

typedef __attribute__((ext_vector_type(16))) __bf16 v16bf;
typedef __attribute__((ext_vector_type(8)))  float  v8f;

union BV16 { v16bf v; __bf16 e[16]; };

__global__ __launch_bounds__(256)
void mixer_toeplitz_wmma(const float* __restrict__ x,
                         const float* __restrict__ weight,
                         const float* __restrict__ bias,
                         float* __restrict__ out)
{
  const int tid  = threadIdx.x;
  const int lane = tid & 31;     // gfx1250 is wave32
  const int wave = tid >> 5;     // 8 waves per workgroup
  const int jt   = blockIdx.x;   // N tile (128 columns)
  const int mt   = blockIdx.y;   // M tile (128 rows)
  const int h    = blockIdx.z;   // head

  const int j0 = jt * 128;

  // LDS: guarded bf16 weight row + double-buffered staged B fragments
  // (2 bufs x 8 frags x 32 lanes x 32B = 16 KB; single barrier per k-step)
  __shared__ __align__(32) __bf16 wlds[GUARD + Ss];
  __shared__ __align__(32) __bf16 bstage[2][8 * 32 * 16];

  for (int i = tid; i < GUARD; i += 256) wlds[i] = (__bf16)0.0f;
  for (int s = tid; s < Ss; s += 256)    wlds[GUARD + s] = (__bf16)weight[h * Ss + s];
  __syncthreads();

  // This wave's 16 A rows. A layout: lanes 0-15 and 16-31 both hold M=0..15.
  const int Rbase = mt * 128 + wave * 16;
  const int arow  = Rbase + (lane & 15);
  const int ab    = arow >> 6;   // batch index
  const int ad    = arow & 63;   // per-head dim index
  const float* aptr = x + ((size_t)ab * Ee + (size_t)h * Dd + ad) * (size_t)Ss;
  const int c0 = (lane >> 4) << 3;   // A K-group base: lanes 0-15 -> 0, lanes 16-31 -> 8

  // Toeplitz generator base for the fragment this wave builds:
  // B[k][j] = w[(j0 + wave*16 + colB) - (k0 + kb + (i&7) + 16*(i>>3))], guard zeros cover j < k.
  const int colB = lane & 15;
  const int kb   = (lane >> 4) << 3;                 // B layout: lanes0-15 K=0..7,16..23, lanes16-31 K=8..15,24..31
  const __bf16* wgen = &wlds[GUARD + j0 + wave * 16 + colB - kb];
  __bf16* bput = &bstage[0][(wave * 32 + lane) * 16];
  const size_t bufstride = 8 * 32 * 16;

  v8f acc[8];
#pragma unroll
  for (int f = 0; f < 8; ++f)
    acc[f] = (v8f){0.f, 0.f, 0.f, 0.f, 0.f, 0.f, 0.f, 0.f};

  const int kend = (j0 + 128 < Ss) ? (j0 + 128) : Ss;
  int buf = 0;

  // ================= main loop: k0 <= j0, all 8 fragments active ================
  for (int k0 = 0; k0 <= j0; k0 += 32, buf ^= 1) {
    // build this wave's B fragment (unconditional: tile fully inside triangle band)
    {
      BV16 bv;
#pragma unroll
      for (int i = 0; i < 16; ++i) bv.e[i] = wgen[-k0 - ((i & 7) + ((i >> 3) << 4))];
      *(v16bf*)(bput + (size_t)buf * bufstride) = bv.v;
    }

    // load + convert this wave's 16x32 A fragment (fp32 -> bf16)
    const float4 a0 = *(const float4*)(aptr + k0 + c0);
    const float4 a1 = *(const float4*)(aptr + k0 + c0 + 4);
    const float4 a2 = *(const float4*)(aptr + k0 + c0 + 16);
    const float4 a3 = *(const float4*)(aptr + k0 + c0 + 20);
    BV16 av;
    av.e[0]  = (__bf16)a0.x; av.e[1]  = (__bf16)a0.y; av.e[2]  = (__bf16)a0.z; av.e[3]  = (__bf16)a0.w;
    av.e[4]  = (__bf16)a1.x; av.e[5]  = (__bf16)a1.y; av.e[6]  = (__bf16)a1.z; av.e[7]  = (__bf16)a1.w;
    av.e[8]  = (__bf16)a2.x; av.e[9]  = (__bf16)a2.y; av.e[10] = (__bf16)a2.z; av.e[11] = (__bf16)a2.w;
    av.e[12] = (__bf16)a3.x; av.e[13] = (__bf16)a3.y; av.e[14] = (__bf16)a3.z; av.e[15] = (__bf16)a3.w;

    __syncthreads();   // fragments of this k-step visible (double-buffer kills WAR barrier)

    // gather all 8 fragments first so DS loads overlap the WMMA chain
    v16bf bfr[8];
#pragma unroll
    for (int f = 0; f < 8; ++f)
      bfr[f] = *(const v16bf*)&bstage[buf][(f * 32 + lane) * 16];

#pragma unroll
    for (int f = 0; f < 8; ++f)
      acc[f] = __builtin_amdgcn_wmma_f32_16x16x32_bf16(
                   false, av.v, false, bfr[f], (short)0, acc[f], false, false);
  }

  // ============ epilogue: diagonal band, k0 = j0+32 .. j0+96 (<=3 iters) ========
  for (int k0 = j0 + 32; k0 < kend; k0 += 32, buf ^= 1) {
    if (k0 <= j0 + wave * 16 + 15) {        // wave-uniform: skip all-zero sub-tiles
      BV16 bv;
#pragma unroll
      for (int i = 0; i < 16; ++i) bv.e[i] = wgen[-k0 - ((i & 7) + ((i >> 3) << 4))];
      *(v16bf*)(bput + (size_t)buf * bufstride) = bv.v;
    }

    const float4 a0 = *(const float4*)(aptr + k0 + c0);
    const float4 a1 = *(const float4*)(aptr + k0 + c0 + 4);
    const float4 a2 = *(const float4*)(aptr + k0 + c0 + 16);
    const float4 a3 = *(const float4*)(aptr + k0 + c0 + 20);
    BV16 av;
    av.e[0]  = (__bf16)a0.x; av.e[1]  = (__bf16)a0.y; av.e[2]  = (__bf16)a0.z; av.e[3]  = (__bf16)a0.w;
    av.e[4]  = (__bf16)a1.x; av.e[5]  = (__bf16)a1.y; av.e[6]  = (__bf16)a1.z; av.e[7]  = (__bf16)a1.w;
    av.e[8]  = (__bf16)a2.x; av.e[9]  = (__bf16)a2.y; av.e[10] = (__bf16)a2.z; av.e[11] = (__bf16)a2.w;
    av.e[12] = (__bf16)a3.x; av.e[13] = (__bf16)a3.y; av.e[14] = (__bf16)a3.z; av.e[15] = (__bf16)a3.w;

    __syncthreads();

#pragma unroll
    for (int f = 0; f < 8; ++f) {
      if (k0 <= j0 + f * 16 + 15) {   // wave-uniform; EXEC stays all-1s for WMMA
        v16bf bv = *(const v16bf*)&bstage[buf][(f * 32 + lane) * 16];
        acc[f] = __builtin_amdgcn_wmma_f32_16x16x32_bf16(
                     false, av.v, false, bv, (short)0, acc[f], false, false);
      }
    }
  }

  // ---- store: C layout lanes 0-15 -> M=v, lanes 16-31 -> M=v+8, N = lane&15 ----
  const int mrow_off = (lane >> 4) << 3;
#pragma unroll
  for (int f = 0; f < 8; ++f) {
    const int colg = j0 + f * 16 + (lane & 15);
    const float bs = bias[h * Ss + colg];
#pragma unroll
    for (int v = 0; v < 8; ++v) {
      const int R  = Rbase + mrow_off + v;
      const int rb = R >> 6;
      const int rd = R & 63;
      out[((size_t)rb * Ee + (size_t)h * Dd + rd) * (size_t)Ss + colg] = acc[f][v] + bs;
    }
  }
}

extern "C" void kernel_launch(void* const* d_in, const int* in_sizes, int n_in,
                              void* d_out, int out_size, void* d_ws, size_t ws_size,
                              hipStream_t stream) {
  const float* x      = (const float*)d_in[0];   // (B, E, S) fp32
  const float* weight = (const float*)d_in[1];   // (H, S)    fp32
  const float* bias   = (const float*)d_in[2];   // (H, S)    fp32
  float* out          = (float*)d_out;           // (B, E, S) fp32

  dim3 grid(Ss / 128, ROWS / 128, Hh);           // (16, 4, 16) workgroups
  mixer_toeplitz_wmma<<<grid, 256, 0, stream>>>(x, weight, bias, out);
}
